// block_cs_ann_77446850281997
// MI455X (gfx1250) — matrix operand order repaired and checked by reading, not compiled, not run on hardware
//
#include <hip/hip_runtime.h>
#include <hip/hip_bf16.h>

// ---------------------------------------------------------------------------
// Problem constants (from the reference)
// ---------------------------------------------------------------------------
static constexpr int B_  = 8;
static constexpr int C_  = 96;
static constexpr int H_  = 96;
static constexpr int W_  = 48;
static constexpr int F_  = 64;          // nb_features
static constexpr int C2_ = C_ / 2;      // 48
static constexpr int H2_ = H_ / 2;      // 48
static constexpr int N_  = C2_ * H2_;   // 2304 tokens

// ---------------------------------------------------------------------------
// WMMA fragment plumbing (gfx1250, wave32, 16x16x32 f16 -> f32)
// ---------------------------------------------------------------------------
typedef __attribute__((ext_vector_type(16))) _Float16 v16h;
typedef __attribute__((ext_vector_type(8)))  _Float16 v8h;
typedef __attribute__((ext_vector_type(8)))  float    v8f;

union AF { v16h v; v8h h8[2]; _Float16 h[16]; };
union CF { v8f  v; float f[8]; };

__device__ __forceinline__ v8f wmma32(v16h a, v16h b, v8f c) {
  return __builtin_amdgcn_wmma_f32_16x16x32_f16(
      /*neg_a=*/false, a, /*neg_b=*/false, b,
      /*c_mod=*/(short)0, c, /*reuse_a=*/false, /*reuse_b=*/false);
}

__device__ __forceinline__ void wait_async0() {
#if __has_builtin(__builtin_amdgcn_s_wait_asynccnt)
  __builtin_amdgcn_s_wait_asynccnt(0);
#else
  asm volatile("s_wait_asynccnt 0x0" ::: "memory");
#endif
}

// Async copy of 16 bytes/lane global -> LDS (GLOBAL_LOAD_ASYNC_TO_LDS_B128,
// tracked by ASYNCcnt).  ldsoff = raw LDS byte offset (low 32 bits of a
// generic pointer into __shared__), gptr = global address.
__device__ __forceinline__ void async_g2l_b128(unsigned ldsoff, const void* gptr) {
  asm volatile("global_load_async_to_lds_b128 %0, %1, off"
               :: "v"(ldsoff), "v"(gptr)
               : "memory");
}

// A fragment of a row-major (M x K) f16 matrix: lane L holds row L&15,
// K = kb..kb+15, kb = 16*(L>>4).
__device__ __forceinline__ v16h load_a16(const _Float16* __restrict__ base,
                                         int row0, int k0, int ld, int lane) {
  int r  = row0 + (lane & 15);
  int kb = (lane >> 4) << 4;
  const _Float16* p = base + (size_t)r * ld + k0 + kb;
  AF a;
  a.h8[0] = *(const v8h*)(p);
  a.h8[1] = *(const v8h*)(p + 8);
  return a.v;
}

// B fragment of B[k][n] (32x16) whose transpose is row-major in memory:
// baseT[n][k].  lane L holds column n0+(L&15), K = kb..kb+15, kb = 16*(L>>4).
__device__ __forceinline__ v16h load_b16(const _Float16* __restrict__ baseT,
                                         int n0, int k0, int ld, int lane) {
  int n  = n0 + (lane & 15);
  int kb = (lane >> 4) << 4;
  const _Float16* p = baseT + (size_t)n * ld + k0 + kb;
  AF b;
  b.h8[0] = *(const v8h*)(p);
  b.h8[1] = *(const v8h*)(p + 8);
  return b.v;
}

// ---------------------------------------------------------------------------
// 1) weight prep: f16 transposed copies of sal_{q,k,v}_w  (64x64 each)
// ---------------------------------------------------------------------------
__global__ void prep_w_kernel(const float* __restrict__ qw,
                              const float* __restrict__ kw,
                              const float* __restrict__ vw,
                              _Float16* __restrict__ wT16) {
  int i = blockIdx.x * 256 + threadIdx.x;
  if (i >= F_ * F_) return;
  int g = i / F_, f = i % F_;
  wT16[0 * F_ * F_ + f * F_ + g] = (_Float16)qw[i];
  wT16[1 * F_ * F_ + f * F_ + g] = (_Float16)kw[i];
  wT16[2 * F_ * F_ + f * F_ + g] = (_Float16)vw[i];
}

// 2) sal_wp (C2,N,H2) -> wpT16[(c*H2+h)][n]  f16   (2304 x 2304)
__global__ void prep_wp_kernel(const float* __restrict__ wp,
                               _Float16* __restrict__ wpT) {
  size_t idx = (size_t)blockIdx.x * 256 + threadIdx.x;
  if (idx >= (size_t)N_ * N_) return;
  int n = (int)(idx % N_);
  int r = (int)(idx / N_);
  int c = r / H2_, hh = r % H2_;
  wpT[idx] = (_Float16)wp[((size_t)c * N_ + n) * H2_ + hh];
}

// ---------------------------------------------------------------------------
// 3) BatchNorm(eval) + SiLU + joint (C,H) 2x2 maxpool -> xp (B,C2,H2,W)
// ---------------------------------------------------------------------------
__global__ void bn_silu_pool_kernel(const float* __restrict__ x,
                                    const float* __restrict__ g,
                                    const float* __restrict__ be,
                                    const float* __restrict__ mu,
                                    const float* __restrict__ var,
                                    float* __restrict__ xp) {
  int idx = blockIdx.x * 256 + threadIdx.x;
  if (idx >= B_ * C2_ * H2_ * W_) return;
  int w = idx % W_;
  int t = idx / W_;
  int h2 = t % H2_; t /= H2_;
  int c2 = t % C2_;
  int b  = t / C2_;
  float best = -1e30f;
#pragma unroll
  for (int dc = 0; dc < 2; ++dc) {
    int c = 2 * c2 + dc;
    float sc = g[c] * rsqrtf(var[c] + 1e-5f);
    float sh = be[c] - mu[c] * sc;
#pragma unroll
    for (int dh = 0; dh < 2; ++dh) {
      int hh = 2 * h2 + dh;
      float v = x[(((size_t)b * C_ + c) * H_ + hh) * W_ + w];
      v = v * sc + sh;
      float s = v / (1.f + __expf(-v));   // SiLU
      best = fmaxf(best, s);
    }
  }
  xp[idx] = best;
}

// ---------------------------------------------------------------------------
// 4) ChannelAttention 1x1 conv over W (one 64-thread block per (b,token) row)
// ---------------------------------------------------------------------------
__global__ void __launch_bounds__(64)
ca_conv_kernel(const float* __restrict__ xp, const float* __restrict__ cw,
               const float* __restrict__ cb, float* __restrict__ y) {
  __shared__ float row[W_];
  int br = blockIdx.x;             // b*N + r
  int f  = threadIdx.x;
  if (f < W_) row[f] = xp[(size_t)br * W_ + f];
  __syncthreads();
  float acc = cb[f];
#pragma unroll
  for (int w = 0; w < W_; ++w) acc += row[w] * cw[f * W_ + w];
  y[(size_t)br * F_ + f] = acc;
}

// ---------------------------------------------------------------------------
// 5) channel gate: max/mean over tokens, two 64x64 GEMVs, softmax -> gate(B,F)
// ---------------------------------------------------------------------------
__global__ void __launch_bounds__(256)
ca_gate_kernel(const float* __restrict__ y,
               const float* __restrict__ l1w, const float* __restrict__ l1b,
               const float* __restrict__ l2w, const float* __restrict__ l2b,
               float* __restrict__ gate) {
  __shared__ float smax[256], ssum[256];
  __shared__ float zmx[F_], zmn[F_], zarr[F_], red[2];
  int b = blockIdx.x, tid = threadIdx.x;
  int f = tid & 63, chunk = tid >> 6;               // 4 chunks of 576 rows
  const float* Y = y + (size_t)b * N_ * F_;
  float mx = -1e30f, sm = 0.f;
  for (int r = chunk * (N_ / 4); r < (chunk + 1) * (N_ / 4); ++r) {
    float v = Y[(size_t)r * F_ + f];
    mx = fmaxf(mx, v);
    sm += v;
  }
  smax[tid] = mx; ssum[tid] = sm;
  __syncthreads();
  if (tid < F_) {
    for (int k = 1; k < 4; ++k) {
      mx = fmaxf(mx, smax[f + 64 * k]);
      sm += ssum[f + 64 * k];
    }
    zmx[f] = mx;
    zmn[f] = sm / (float)N_;
  }
  __syncthreads();
  if (tid < F_) {
    float z = l1b[f] + l2b[f];
    for (int g = 0; g < F_; ++g)
      z += zmx[g] * l1w[g * F_ + f] + zmn[g] * l2w[g * F_ + f];
    zarr[f] = z;
  }
  __syncthreads();
  if (tid == 0) {
    float m = zarr[0];
    for (int i = 1; i < F_; ++i) m = fmaxf(m, zarr[i]);
    red[0] = m;
  }
  __syncthreads();
  if (tid < F_) zarr[f] = __expf(zarr[f] - red[0]);
  __syncthreads();
  if (tid == 0) {
    float s = 0.f;
    for (int i = 0; i < F_; ++i) s += zarr[i];
    red[1] = s;
  }
  __syncthreads();
  if (tid < F_) gate[b * F_ + f] = zarr[f] / red[1];
}

// 6) x_ca = y * gate  (f32 + f16 copies)
__global__ void apply_gate_kernel(const float* __restrict__ y,
                                  const float* __restrict__ gate,
                                  float* __restrict__ xca,
                                  _Float16* __restrict__ xca16) {
  int idx = blockIdx.x * 256 + threadIdx.x;
  if (idx >= B_ * N_ * F_) return;
  int f = idx & 63;
  int b = idx / (N_ * F_);
  float v = y[idx] * gate[b * F_ + f];
  xca[idx]   = v;
  xca16[idx] = (_Float16)v;
}

// ---------------------------------------------------------------------------
// 7) q/k/v projections: (2304x64)@(64x64)+bias via WMMA.
//    All 8 weight B-fragments per projection are loaded up-front into
//    distinct registers (one clause), so WMMAs don't drain loadcnt to 0
//    one by one.
// ---------------------------------------------------------------------------
__global__ void __launch_bounds__(128)
qkv_proj_kernel(const _Float16* __restrict__ xca16,
                const _Float16* __restrict__ wT16,
                const float* __restrict__ qb, const float* __restrict__ kb,
                const float* __restrict__ vb,
                _Float16* __restrict__ q16, _Float16* __restrict__ k16,
                _Float16* __restrict__ vT16) {
  int lane = threadIdx.x & 31, wave = threadIdx.x >> 5;
  int b = blockIdx.y;
  int row0 = blockIdx.x * 64 + wave * 16;
  int h = lane >> 4, nl = lane & 15;
  const _Float16* X = xca16 + (size_t)b * N_ * F_;

  v16h a0 = load_a16(X, row0, 0,  F_, lane);
  v16h a1 = load_a16(X, row0, 32, F_, lane);

  const float* biases[3] = {qb, kb, vb};
  for (int p = 0; p < 3; ++p) {
    const _Float16* Wt = wT16 + p * F_ * F_;
    const float* bias = biases[p];

    // batch-load all 8 weight fragments for this projection
    v16h bw[8];
#pragma unroll
    for (int ff = 0; ff < 4; ++ff) {
      bw[2 * ff]     = load_b16(Wt, 16 * ff, 0,  F_, lane);
      bw[2 * ff + 1] = load_b16(Wt, 16 * ff, 32, F_, lane);
    }

#pragma unroll
    for (int ff = 0; ff < 4; ++ff) {
      CF c;
#pragma unroll
      for (int j = 0; j < 8; ++j) c.f[j] = 0.f;
      c.v = wmma32(a0, bw[2 * ff],     c.v);
      c.v = wmma32(a1, bw[2 * ff + 1], c.v);
      float bf = bias[16 * ff + nl];
#pragma unroll
      for (int j = 0; j < 8; ++j) {
        float val = c.f[j] + bf;
        int r = row0 + j + 8 * h;
        if (p == 0)
          q16[(size_t)b * N_ * F_ + (size_t)r * F_ + 16 * ff + nl] = (_Float16)val;
        else if (p == 1)
          k16[(size_t)b * N_ * F_ + (size_t)r * F_ + 16 * ff + nl] = (_Float16)val;
        else
          vT16[(size_t)b * F_ * N_ + (size_t)(16 * ff + nl) * N_ + r] = (_Float16)val;
      }
    }
  }
}

// ---------------------------------------------------------------------------
// 8) Flash attention with async-LDS staging:
//    - per block: 4 waves x 16-query tiles (64 queries)
//    - per 32-key iteration: K-tile (32x64 f16) and V-tile (64x32 f16, from
//      vT) staged ONCE into LDS via GLOBAL_LOAD_ASYNC_TO_LDS_B128 (ASYNCcnt),
//      double buffered so tile i+1 streams while tile i feeds the WMMAs.
//    - online softmax, P transposed C->A layout through LDS.
// ---------------------------------------------------------------------------
__global__ void __launch_bounds__(128)
flash_attn_kernel(const _Float16* __restrict__ q16,
                  const _Float16* __restrict__ k16,
                  const _Float16* __restrict__ vT16,
                  _Float16* __restrict__ qkvT16) {
  __shared__ _Float16 kbuf[2][32 * 64];     // keys x features (rows of 64)
  __shared__ _Float16 vbuf[2][64 * 32];     // features x keys (rows of 32)
  __shared__ _Float16 pl[4][16][40];        // per-wave P tile (16x32, padded)

  int tid = threadIdx.x;
  int lane = tid & 31, wave = tid >> 5;
  int b = blockIdx.y;
  int q0 = blockIdx.x * 64 + wave * 16;
  int h = lane >> 4, nl = lane & 15;

  const _Float16* Q = q16    + (size_t)b * N_ * F_;
  const _Float16* K = k16    + (size_t)b * N_ * F_;
  const _Float16* V = vT16   + (size_t)b * F_ * N_;
  _Float16*       O = qkvT16 + (size_t)b * F_ * N_;

  // stage one 32-key tile (K: 4KB contiguous; V: 64 rows x 64B strided)
  auto stage = [&](int kv, int bi) {
#pragma unroll
    for (int r2 = 0; r2 < 2; ++r2) {
      int seg = r2 * 128 + tid;                       // 0..255, 16B segments
      const _Float16* gk = K + (size_t)kv * F_ + seg * 8;
      async_g2l_b128((unsigned)(size_t)(&kbuf[bi][seg * 8]), gk);
      int row = seg >> 2, j = seg & 3;
      const _Float16* gv = V + (size_t)row * N_ + kv + j * 8;
      async_g2l_b128((unsigned)(size_t)(&vbuf[bi][seg * 8]), gv);
    }
  };
  // B fragment of K^T from LDS K-tile (feat-contraction, key columns)
  auto frag_k = [&](int bi, int n0, int k0) -> v16h {
    AF bf;
    const _Float16* p = &kbuf[bi][(n0 + nl) * 64 + k0 + h * 16];
    bf.h8[0] = *(const v8h*)(p);
    bf.h8[1] = *(const v8h*)(p + 8);
    return bf.v;
  };
  // B fragment of V from LDS V-tile (key-contraction, feature columns)
  auto frag_v = [&](int bi, int f0) -> v16h {
    AF bf;
    const _Float16* p = &vbuf[bi][(f0 + nl) * 32 + h * 16];
    bf.h8[0] = *(const v8h*)(p);
    bf.h8[1] = *(const v8h*)(p + 8);
    return bf.v;
  };

  v16h qa0 = load_a16(Q, q0, 0,  F_, lane);
  v16h qa1 = load_a16(Q, q0, 32, F_, lane);

  CF o[4];
  float mrow[8], lrow[8];
#pragma unroll
  for (int ff = 0; ff < 4; ++ff)
#pragma unroll
    for (int j = 0; j < 8; ++j) o[ff].f[j] = 0.f;
#pragma unroll
  for (int j = 0; j < 8; ++j) { mrow[j] = -1e30f; lrow[j] = 0.f; }

  stage(0, 0);
  for (int kv = 0, it = 0; kv < N_; kv += 32, ++it) {
    int bi = it & 1;
    wait_async0();          // this wave's async copies for tile `it` landed
    __syncthreads();        // everyone's copies landed

    // pull this tile's fragments out of LDS (ds_load_b128s)
    v16h kb00 = frag_k(bi, 0,  0);
    v16h kb01 = frag_k(bi, 0,  32);
    v16h kb10 = frag_k(bi, 16, 0);
    v16h kb11 = frag_k(bi, 16, 32);
    v16h vb0 = frag_v(bi, 0);
    v16h vb1 = frag_v(bi, 16);
    v16h vb2 = frag_v(bi, 32);
    v16h vb3 = frag_v(bi, 48);

    // stream next tile into the other buffer while we compute
    if (kv + 32 < N_) stage(kv + 32, bi ^ 1);

    // S = Q K^T (two 16-col fragments), contraction over F=64
    CF s0, s1;
#pragma unroll
    for (int j = 0; j < 8; ++j) { s0.f[j] = 0.f; s1.f[j] = 0.f; }
    s0.v = wmma32(qa0, kb00, s0.v);
    s0.v = wmma32(qa1, kb01, s0.v);
    s1.v = wmma32(qa0, kb10, s1.v);
    s1.v = wmma32(qa1, kb11, s1.v);

    // online softmax (row stats; reduce over the 16-lane half)
    float p0a[8], p1a[8];
#pragma unroll
    for (int j = 0; j < 8; ++j) {
      float mj = fmaxf(s0.f[j], s1.f[j]);
#pragma unroll
      for (int off = 8; off >= 1; off >>= 1)
        mj = fmaxf(mj, __shfl_xor(mj, off, 32));
      float mn = fmaxf(mrow[j], mj);
      float p0 = __expf(s0.f[j] - mn);
      float p1 = __expf(s1.f[j] - mn);
      float rs = p0 + p1;
#pragma unroll
      for (int off = 8; off >= 1; off >>= 1) rs += __shfl_xor(rs, off, 32);
      float corr = __expf(mrow[j] - mn);
      lrow[j] = lrow[j] * corr + rs;
      mrow[j] = mn;
      p0a[j] = p0; p1a[j] = p1;
#pragma unroll
      for (int ff = 0; ff < 4; ++ff) o[ff].f[j] *= corr;
    }

    // transpose P (C layout -> A layout) via LDS
#pragma unroll
    for (int j = 0; j < 8; ++j) {
      pl[wave][j + 8 * h][nl]      = (_Float16)p0a[j];
      pl[wave][j + 8 * h][16 + nl] = (_Float16)p1a[j];
    }
    __syncthreads();
    AF pa;
#pragma unroll
    for (int i = 0; i < 8; ++i) {
      pa.h[i]     = pl[wave][nl][16 * h + i];
      pa.h[8 + i] = pl[wave][nl][16 * h + 8 + i];
    }
    __syncthreads();

    // O += P @ V
    o[0].v = wmma32(pa.v, vb0, o[0].v);
    o[1].v = wmma32(pa.v, vb1, o[1].v);
    o[2].v = wmma32(pa.v, vb2, o[2].v);
    o[3].v = wmma32(pa.v, vb3, o[3].v);
  }

  // normalize and store transposed f16
#pragma unroll
  for (int ff = 0; ff < 4; ++ff)
#pragma unroll
    for (int j = 0; j < 8; ++j) {
      float val = o[ff].f[j] / lrow[j];
      O[(size_t)(16 * ff + nl) * N_ + q0 + j + 8 * h] = (_Float16)val;
    }
}

// ---------------------------------------------------------------------------
// 9) SpatialAttentionConvolution: tiny 48x48 attention per batch, scalar+LDS.
// ---------------------------------------------------------------------------
__global__ void __launch_bounds__(256)
sac_attn_kernel(const float* __restrict__ xca,
                const float* __restrict__ qw, const float* __restrict__ qb,
                const float* __restrict__ kw, const float* __restrict__ kb,
                const float* __restrict__ vw, const float* __restrict__ vb,
                float* __restrict__ qkv_c) {
  __shared__ float sq[H2_ * F_], sk[H2_ * F_], sv[H2_ * F_];
  __shared__ float att[H2_ * H2_];
  int b = blockIdx.x, tid = threadIdx.x;
  const float* X = xca + (size_t)b * N_ * F_;

  for (int idx = tid; idx < H2_ * F_; idx += 256) {
    int hh = idx / F_, f = idx % F_;
    float aq = qb[0], ak = kb[0], av = vb[0];
    for (int c = 0; c < C2_; ++c) {
      float xv = X[(size_t)(c * H2_ + hh) * F_ + f];
      aq += xv * qw[c]; ak += xv * kw[c]; av += xv * vw[c];
    }
    sq[idx] = aq; sk[idx] = ak; sv[idx] = av;
  }
  __syncthreads();
  for (int idx = tid; idx < H2_ * H2_; idx += 256) {
    int hh = idx / H2_, gg = idx % H2_;
    float s = 0.f;
    for (int f = 0; f < F_; ++f) s += sq[hh * F_ + f] * sk[gg * F_ + f];
    att[idx] = s;
  }
  __syncthreads();
  if (tid < H2_) {
    float m = -1e30f;
    for (int g = 0; g < H2_; ++g) m = fmaxf(m, att[tid * H2_ + g]);
    float s = 0.f;
    for (int g = 0; g < H2_; ++g) {
      float e = __expf(att[tid * H2_ + g] - m);
      att[tid * H2_ + g] = e;
      s += e;
    }
    float inv = 1.f / s;
    for (int g = 0; g < H2_; ++g) att[tid * H2_ + g] *= inv;
  }
  __syncthreads();
  for (int idx = tid; idx < H2_ * F_; idx += 256) {
    int hh = idx / F_, f = idx % F_;
    float oacc = 0.f;
    for (int g = 0; g < H2_; ++g) oacc += att[hh * H2_ + g] * sv[g * F_ + f];
    qkv_c[(size_t)b * H2_ * F_ + idx] = oacc;
  }
}

// ---------------------------------------------------------------------------
// 10) z_lin GEMM (wpT @ qkv, K=2304) + fused z_conv + final * x_ca.
//     Register double-buffering: iteration i+1's fragments are in flight
//     while iteration i's WMMAs execute.
// ---------------------------------------------------------------------------
__global__ void __launch_bounds__(128)
zlin_out_kernel(const _Float16* __restrict__ wpT16,
                const _Float16* __restrict__ qkvT16,
                const float* __restrict__ qkv_c,
                const float* __restrict__ sac_o_w,
                const float* __restrict__ sac_o_b,
                const float* __restrict__ xca,
                float* __restrict__ out) {
  int lane = threadIdx.x & 31, wave = threadIdx.x >> 5;
  int b = blockIdx.y;
  int row0 = blockIdx.x * 64 + wave * 16;
  int h = lane >> 4, nl = lane & 15;
  const _Float16* Qk = qkvT16 + (size_t)b * F_ * N_;

  CF acc[4];
#pragma unroll
  for (int ff = 0; ff < 4; ++ff)
#pragma unroll
    for (int j = 0; j < 8; ++j) acc[ff].f[j] = 0.f;

  v16h a_cur = load_a16(wpT16, row0, 0, N_, lane);
  v16h b_cur[4];
#pragma unroll
  for (int ff = 0; ff < 4; ++ff) b_cur[ff] = load_b16(Qk, 16 * ff, 0, N_, lane);

  for (int k0 = 0; k0 < N_; k0 += 32) {
    v16h a_nxt = a_cur;
    v16h b_nxt[4] = {b_cur[0], b_cur[1], b_cur[2], b_cur[3]};
    int kn = k0 + 32;
    if (kn < N_) {
      // prefetch the HBM-streamed wpT line one more iteration ahead
      __builtin_prefetch(wpT16 + (size_t)(row0 + (lane & 15)) * N_ + kn + 32, 0, 1);
      a_nxt = load_a16(wpT16, row0, kn, N_, lane);
#pragma unroll
      for (int ff = 0; ff < 4; ++ff) b_nxt[ff] = load_b16(Qk, 16 * ff, kn, N_, lane);
    }
#pragma unroll
    for (int ff = 0; ff < 4; ++ff) acc[ff].v = wmma32(a_cur, b_cur[ff], acc[ff].v);
    a_cur = a_nxt;
#pragma unroll
    for (int ff = 0; ff < 4; ++ff) b_cur[ff] = b_nxt[ff];
  }

#pragma unroll
  for (int ff = 0; ff < 4; ++ff)
#pragma unroll
    for (int j = 0; j < 8; ++j) {
      int r = row0 + j + 8 * h;
      int cc = r / H2_, hh = r % H2_;
      int f = 16 * ff + nl;
      float val = acc[ff].f[j]
                + qkv_c[(size_t)b * H2_ * F_ + hh * F_ + f] * sac_o_w[cc]
                + sac_o_b[cc];
      size_t oi = ((size_t)b * N_ + r) * F_ + f;
      out[oi] = val * xca[oi];
    }
}

// ---------------------------------------------------------------------------
// launcher
// ---------------------------------------------------------------------------
extern "C" void kernel_launch(void* const* d_in, const int* in_sizes, int n_in,
                              void* d_out, int out_size, void* d_ws, size_t ws_size,
                              hipStream_t stream) {
  (void)in_sizes; (void)n_in; (void)out_size; (void)ws_size;
  const float* x        = (const float*)d_in[0];
  const float* bn_gamma = (const float*)d_in[1];
  const float* bn_beta  = (const float*)d_in[2];
  const float* bn_mean  = (const float*)d_in[3];
  const float* bn_var   = (const float*)d_in[4];
  const float* ca_conv_w = (const float*)d_in[5];
  const float* ca_conv_b = (const float*)d_in[6];
  const float* ca_l1_w = (const float*)d_in[7];
  const float* ca_l1_b = (const float*)d_in[8];
  const float* ca_l2_w = (const float*)d_in[9];
  const float* ca_l2_b = (const float*)d_in[10];
  const float* sal_q_w = (const float*)d_in[11];
  const float* sal_q_b = (const float*)d_in[12];
  const float* sal_k_w = (const float*)d_in[13];
  const float* sal_k_b = (const float*)d_in[14];
  const float* sal_v_w = (const float*)d_in[15];
  const float* sal_v_b = (const float*)d_in[16];
  const float* sal_wp  = (const float*)d_in[17];
  const float* sac_q_w = (const float*)d_in[18];
  const float* sac_q_b = (const float*)d_in[19];
  const float* sac_k_w = (const float*)d_in[20];
  const float* sac_k_b = (const float*)d_in[21];
  const float* sac_v_w = (const float*)d_in[22];
  const float* sac_v_b = (const float*)d_in[23];
  const float* sac_o_w = (const float*)d_in[24];
  const float* sac_o_b = (const float*)d_in[25];
  float* out = (float*)d_out;

  // workspace carve-up (256B aligned)
  char* wsb = (char*)d_ws;
  size_t off = 0;
  auto carve = [&](size_t bytes) -> void* {
    void* p = wsb + off;
    off = (off + bytes + 255) & ~size_t(255);
    return p;
  };
  float*    xp     = (float*)   carve((size_t)B_ * N_ * W_ * 4);
  float*    y      = (float*)   carve((size_t)B_ * N_ * F_ * 4);
  float*    xca    = (float*)   carve((size_t)B_ * N_ * F_ * 4);
  float*    gate   = (float*)   carve((size_t)B_ * F_ * 4);
  _Float16* xca16  = (_Float16*)carve((size_t)B_ * N_ * F_ * 2);
  _Float16* wT16   = (_Float16*)carve((size_t)3 * F_ * F_ * 2);
  _Float16* q16    = (_Float16*)carve((size_t)B_ * N_ * F_ * 2);
  _Float16* k16    = (_Float16*)carve((size_t)B_ * N_ * F_ * 2);
  _Float16* vT16   = (_Float16*)carve((size_t)B_ * N_ * F_ * 2);
  _Float16* qkvT16 = (_Float16*)carve((size_t)B_ * N_ * F_ * 2);
  _Float16* wpT16  = (_Float16*)carve((size_t)N_ * N_ * 2);
  float*    qkvc   = (float*)   carve((size_t)B_ * H2_ * F_ * 4);

  prep_w_kernel<<<(F_ * F_ + 255) / 256, 256, 0, stream>>>(sal_q_w, sal_k_w, sal_v_w, wT16);
  prep_wp_kernel<<<(int)(((size_t)N_ * N_ + 255) / 256), 256, 0, stream>>>(sal_wp, wpT16);

  bn_silu_pool_kernel<<<(B_ * N_ * W_ + 255) / 256, 256, 0, stream>>>(
      x, bn_gamma, bn_beta, bn_mean, bn_var, xp);

  ca_conv_kernel<<<B_ * N_, 64, 0, stream>>>(xp, ca_conv_w, ca_conv_b, y);

  ca_gate_kernel<<<B_, 256, 0, stream>>>(y, ca_l1_w, ca_l1_b, ca_l2_w, ca_l2_b, gate);

  apply_gate_kernel<<<(B_ * N_ * F_ + 255) / 256, 256, 0, stream>>>(y, gate, xca, xca16);

  qkv_proj_kernel<<<dim3(N_ / 64, B_), 128, 0, stream>>>(
      xca16, wT16, sal_q_b, sal_k_b, sal_v_b, q16, k16, vT16);

  flash_attn_kernel<<<dim3(N_ / 64, B_), 128, 0, stream>>>(q16, k16, vT16, qkvT16);

  sac_attn_kernel<<<B_, 256, 0, stream>>>(
      xca, sac_q_w, sac_q_b, sac_k_w, sac_k_b, sac_v_w, sac_v_b, qkvc);

  zlin_out_kernel<<<dim3(N_ / 64, B_), 128, 0, stream>>>(
      wpT16, qkvT16, qkvc, sac_o_w, sac_o_b, xca, out);
}
